// Attention_2611340116117
// MI455X (gfx1250) — compile-verified, matrix operand order checked by reading, not run on hardware
//
#include <hip/hip_runtime.h>
#include <hip/hip_bf16.h>
#include <math.h>

// Attention (causal, single head): B=8, T=4096, C=1024, H=128, fp32 in/out.
// Strategy: f16 WMMA (v_wmma_f32_16x16x32_f16) everywhere, f32 accumulate.
// Workspace (f16): q[B][T][H], k[B][T][H], vT[B][H][T], WqT/WkT/WvT[H][C]
//   = 3*8MB + 3*256KB ~= 24.8 MB.

typedef _Float16 h16;
typedef __attribute__((ext_vector_type(16))) _Float16 v16h;
typedef __attribute__((ext_vector_type(8)))  _Float16 v8h;
typedef __attribute__((ext_vector_type(8)))  float    v8f;

#define B_  8
#define T_  4096
#define C_  1024
#define H_  128

__device__ __forceinline__ v8f zero_v8f() {
    v8f z;
#pragma unroll
    for (int i = 0; i < 8; ++i) z[i] = 0.0f;
    return z;
}

// D = A*B + C  (16x16x32, f16 in, f32 acc)
__device__ __forceinline__ v8f wmma_f16(v16h a, v16h b, v8f c) {
    return __builtin_amdgcn_wmma_f32_16x16x32_f16(
        /*neg_a=*/false, a, /*neg_b=*/false, b,
        /*c_mod=*/(short)0, c, /*reuse_a=*/false, /*reuse_b=*/false);
}

// Load a 16x32 f16 WMMA operand (A or B; identical lane mapping) from a
// row-major f16 matrix. Lane l -> row (l&15); k-positions (l>>4)*8+[0,8)
// and 16+(l>>4)*8+[0,8): two contiguous 16-byte loads.
__device__ __forceinline__ v16h load_op16(const h16* __restrict__ base, int ld, int k0) {
    const int lane = threadIdx.x & 31;
    const int row  = lane & 15;
    const int hh   = lane >> 4;
    const h16* p = base + (size_t)row * ld + k0 + hh * 8;
    v8h lo = *(const v8h*)(p);        // k = k0 + hh*8 + [0,8)
    v8h hi = *(const v8h*)(p + 16);   // k = k0 + 16 + hh*8 + [0,8)
    v16h r;
#pragma unroll
    for (int i = 0; i < 8; ++i) { r[i] = lo[i]; r[8 + i] = hi[i]; }
    return r;
}

// ---------------- Kernel 0: W [C][H] fp32 -> Wt [H][C] f16 ----------------
__global__ void wcvt_kernel(const float* __restrict__ W, h16* __restrict__ Wt) {
    int idx = blockIdx.x * blockDim.x + threadIdx.x;   // over H_*C_
    if (idx >= H_ * C_) return;
    int h = idx >> 10;          // / C_
    int c = idx & (C_ - 1);
    Wt[idx] = (h16)W[(size_t)c * H_ + h];
}

// ---------------- Kernel 1: projections q,k,v ----------------
// One block = one (b, 16-row tile of T). x tile staged in LDS as f16 once;
// 8 waves each own a 16-wide slice of H and chain 32 WMMAs over C=1024.
#define XS_LD 1032   // 1024 + 8 halves pad: kills LDS bank conflicts, keeps 16B align

__global__ __launch_bounds__(256) void proj_kernel(
    const float* __restrict__ x,
    const h16* __restrict__ WqT, const h16* __restrict__ WkT, const h16* __restrict__ WvT,
    h16* __restrict__ qf, h16* __restrict__ kf, h16* __restrict__ vT) {
    __shared__ h16 xs[16 * XS_LD];   // ~33 KB of 320 KB/WGP

    const int tid = threadIdx.x;
    const int b   = blockIdx.x >> 8;          // / (T_/16)
    const int t0  = (blockIdx.x & 255) << 4;

    // cooperative fp32->f16 stage of the 16x1024 x tile
    const float4* xsrc = (const float4*)(x + ((size_t)b * T_ + t0) * C_);
    for (int i = tid; i < 16 * (C_ / 4); i += 256) {
        float4 f = xsrc[i];
        int row = i >> 8;                 // 256 float4 per row
        int c4  = (i & 255) << 2;
        h16* d = xs + row * XS_LD + c4;
        d[0] = (h16)f.x; d[1] = (h16)f.y; d[2] = (h16)f.z; d[3] = (h16)f.w;
    }
    __syncthreads();

    const int wave = tid >> 5;
    const int lane = tid & 31;
    const int n    = lane & 15;
    const int hh   = lane >> 4;
    const int h0   = wave << 4;

    v8f aq = zero_v8f(), ak = zero_v8f(), av = zero_v8f();
    for (int k0 = 0; k0 < C_; k0 += 32) {
        v16h A  = load_op16(xs, XS_LD, k0);                    // ds_load x tile
        v16h bq = load_op16(WqT + (size_t)h0 * C_, C_, k0);    // contiguous B ops
        v16h bk = load_op16(WkT + (size_t)h0 * C_, C_, k0);
        v16h bv = load_op16(WvT + (size_t)h0 * C_, C_, k0);
        aq = wmma_f16(A, bq, aq);
        ak = wmma_f16(A, bk, ak);
        av = wmma_f16(A, bv, av);
    }

    // q,k row-major [T][H]; C-layout: VGPR r -> row r+8*hh, col n
    size_t qkbase = ((size_t)b * T_ + t0) * H_ + h0 + n;
#pragma unroll
    for (int r = 0; r < 8; ++r) {
        size_t off = qkbase + (size_t)(r + 8 * hh) * H_;
        qf[off] = (h16)aq[r];
        kf[off] = (h16)ak[r];
    }
    // v transposed [H][T]: per lane the 8 accumulator rows are 8 consecutive
    // t positions -> single packed 16B store.
    v8h pv;
#pragma unroll
    for (int r = 0; r < 8; ++r) pv[r] = (h16)av[r];
    *(v8h*)(vT + ((size_t)b * H_ + h0 + n) * T_ + t0 + 8 * hh) = pv;
}

// ---------------- Kernel 2: flash attention ----------------
// One wave per 16-query tile; causal loop over 32-key pairs; online softmax.
#define PLD 40   // padded LDS row stride (halves) for the P transpose buffer

__global__ __launch_bounds__(256) void attn_kernel(
    const h16* __restrict__ qf, const h16* __restrict__ kf,
    const h16* __restrict__ vT, float* __restrict__ out) {
    __shared__ h16 pbuf[8 * 16 * PLD];   // 10 KB, one 16x32 tile per wave

    const int tid  = threadIdx.x;
    const int wave = tid >> 5;
    const int lane = tid & 31;
    const int n    = lane & 15;
    const int hh   = lane >> 4;
    const int b    = blockIdx.x >> 5;                 // 32 blocks per batch
    const int qt   = ((blockIdx.x & 31) << 3) + wave; // 0..255
    const int q0   = qt << 4;

    const h16* qb = qf + ((size_t)b * T_ + q0) * H_;
    const h16* kb = kf + (size_t)b * T_ * H_;
    const h16* vb = vT + (size_t)b * H_ * T_;
    h16* pb = pbuf + wave * 16 * PLD;

    v16h Q[4];
#pragma unroll
    for (int c = 0; c < 4; ++c) Q[c] = load_op16(qb, H_, c * 32);

    v8f O[8];
#pragma unroll
    for (int t = 0; t < 8; ++t) O[t] = zero_v8f();
    float mrow[8], lrow[8];
#pragma unroll
    for (int r = 0; r < 8; ++r) { mrow[r] = -INFINITY; lrow[r] = 0.0f; }

    const float scale = 0.08838834764831845f;   // H^-0.5
    const int npairs = (qt >> 1) + 1;

    for (int p = 0; p < npairs; ++p) {
        const int k32 = p << 5;
        if (p + 1 < npairs)
            __builtin_prefetch(kb + (size_t)(k32 + 32) * H_, 0, 0);

        v8f S0, S1;
        {   // key subtile 0: always at least partially valid
            const int kt = k32;
            v8f acc = zero_v8f();
#pragma unroll
            for (int c = 0; c < 4; ++c)
                acc = wmma_f16(Q[c], load_op16(kb + (size_t)kt * H_, H_, c * 32), acc);
#pragma unroll
            for (int r = 0; r < 8; ++r) {
                int query = q0 + r + 8 * hh, key = kt + n;
                S0[r] = (key <= query) ? acc[r] * scale : -INFINITY;
            }
        }
        {   // key subtile 1: may be fully masked on the last pair
            const int kt = k32 + 16;
            if (kt <= q0 + 15) {
                v8f acc = zero_v8f();
#pragma unroll
                for (int c = 0; c < 4; ++c)
                    acc = wmma_f16(Q[c], load_op16(kb + (size_t)kt * H_, H_, c * 32), acc);
#pragma unroll
                for (int r = 0; r < 8; ++r) {
                    int query = q0 + r + 8 * hh, key = kt + n;
                    S1[r] = (key <= query) ? acc[r] * scale : -INFINITY;
                }
            } else {
#pragma unroll
                for (int r = 0; r < 8; ++r) S1[r] = -INFINITY;
            }
        }

        // online softmax; row r+8*hh lives across the 16 lanes of a half-wave
        float aarr[8];
#pragma unroll
        for (int r = 0; r < 8; ++r) {
            float mx = fmaxf(S0[r], S1[r]);
            mx = fmaxf(mx, __shfl_xor(mx, 1));
            mx = fmaxf(mx, __shfl_xor(mx, 2));
            mx = fmaxf(mx, __shfl_xor(mx, 4));
            mx = fmaxf(mx, __shfl_xor(mx, 8));
            float mnew = fmaxf(mrow[r], mx);
            float a  = __expf(mrow[r] - mnew);
            float p0 = __expf(S0[r] - mnew);
            float p1 = __expf(S1[r] - mnew);
            S0[r] = p0; S1[r] = p1;
            float sm = p0 + p1;
            sm += __shfl_xor(sm, 1);
            sm += __shfl_xor(sm, 2);
            sm += __shfl_xor(sm, 4);
            sm += __shfl_xor(sm, 8);
            lrow[r] = lrow[r] * a + sm;
            mrow[r] = mnew;
            aarr[r] = a;
        }
#pragma unroll
        for (int t = 0; t < 8; ++t)
#pragma unroll
            for (int r = 0; r < 8; ++r) O[t][r] *= aarr[r];

        // C-layout -> A-layout via wave-private LDS (DS ops are in-order per wave)
#pragma unroll
        for (int r = 0; r < 8; ++r) {
            int row = r + 8 * hh;
            pb[row * PLD + n]      = (h16)S0[r];
            pb[row * PLD + 16 + n] = (h16)S1[r];
        }
        __builtin_amdgcn_wave_barrier();
        v16h P = load_op16(pb, PLD, 0);
        __builtin_amdgcn_wave_barrier();

        // O += P * V  over the 8 head slices (V read from vT: contiguous)
#pragma unroll
        for (int t = 0; t < 8; ++t) {
            v16h V = load_op16(vb + (size_t)(t * 16) * T_, T_, k32);
            O[t] = wmma_f16(P, V, O[t]);
        }
    }

    // epilogue: normalize and store fp32 [B][T][H]
    float inv[8];
#pragma unroll
    for (int r = 0; r < 8; ++r) inv[r] = 1.0f / lrow[r];
    float* ob = out + ((size_t)b * T_ + q0) * H_;
#pragma unroll
    for (int t = 0; t < 8; ++t)
#pragma unroll
        for (int r = 0; r < 8; ++r)
            ob[(size_t)(r + 8 * hh) * H_ + t * 16 + n] = O[t][r] * inv[r];
}

// ---------------- host launch ----------------
extern "C" void kernel_launch(void* const* d_in, const int* in_sizes, int n_in,
                              void* d_out, int out_size, void* d_ws, size_t ws_size,
                              hipStream_t stream) {
    (void)in_sizes; (void)n_in; (void)out_size; (void)ws_size;
    const float* x  = (const float*)d_in[0];
    const float* Wk = (const float*)d_in[1];   // dict order: x, Wk, Wq, Wv
    const float* Wq = (const float*)d_in[2];
    const float* Wv = (const float*)d_in[3];
    float* out = (float*)d_out;

    h16* ws = (h16*)d_ws;
    const size_t QKV = (size_t)B_ * T_ * H_;
    h16* qf  = ws;
    h16* kf  = qf + QKV;
    h16* vT  = kf + QKV;
    h16* WqT = vT + QKV;
    h16* WkT = WqT + (size_t)H_ * C_;
    h16* WvT = WkT + (size_t)H_ * C_;

    wcvt_kernel<<<(H_ * C_ + 255) / 256, 256, 0, stream>>>(Wq, WqT);
    wcvt_kernel<<<(H_ * C_ + 255) / 256, 256, 0, stream>>>(Wk, WkT);
    wcvt_kernel<<<(H_ * C_ + 255) / 256, 256, 0, stream>>>(Wv, WvT);

    proj_kernel<<<B_ * (T_ / 16), 256, 0, stream>>>(x, WqT, WkT, WvT, qf, kf, vT);
    attn_kernel<<<B_ * (T_ / 16) / 8, 256, 0, stream>>>(qf, kf, vT, out);
}
